// HybridTwoTower_65506841198654
// MI455X (gfx1250) — compile-verified, matrix operand order checked by reading, not run on hardware
//
#include <hip/hip_runtime.h>

#define GCN 64
#define HID 256
#define TOW 128
#define K1I 96   // 66 padded to 3*32
#define K1U 192  // 6*32

typedef __attribute__((ext_vector_type(16))) _Float16     v16h;
typedef __attribute__((ext_vector_type(8)))  float        v8f;
typedef __attribute__((ext_vector_type(4)))  unsigned int v4u;
typedef __attribute__((ext_vector_type(8)))  unsigned int v8u;

union ABFrag { v16h v; unsigned int u[8]; };

__device__ __forceinline__ unsigned short f2h(float f) {
  union { _Float16 h; unsigned short u; } p;
  p.h = (_Float16)f;                       // native v_cvt_f16_f32 (RNE)
  return p.u;
}
__device__ __forceinline__ unsigned int pk2h(float lo, float hi) {
  union { _Float16 h[2]; unsigned int u; } p;
  p.h[0] = (_Float16)lo; p.h[1] = (_Float16)hi;
  return p.u;
}

// ---------------------------------------------------------------------------
// TDM: 1-D contiguous global -> LDS DMA (nbytes % 8 == 0, nbytes/8 <= 65535).
// D# per CDNA5 ISA ch.8: group0 {count, lds_addr, global_addr, type=2},
// group1 {data_size=8B, tensor_dim0=tile_dim0=nqw, tensor_dim1=tile_dim1=1}.
// Issue once per wave; completion tracked with TENSORcnt.
// ---------------------------------------------------------------------------
__device__ __forceinline__ void tdm_load_1d(void* lds_dst, const void* gsrc,
                                            unsigned int nbytes) {
  unsigned int nqw = nbytes >> 3;                                   // 8-byte units
  unsigned long long ga = (unsigned long long)(uintptr_t)gsrc;
  unsigned int lds = (unsigned int)(uintptr_t)lds_dst;              // flat low32 = LDS offset
  v4u g0;
  g0[0] = 1u;                                                       // count=1, no gather
  g0[1] = lds;                                                      // lds_addr
  g0[2] = (unsigned int)ga;                                         // global_addr[31:0]
  g0[3] = (unsigned int)(ga >> 32) | 0x80000000u;                   // addr[56:32] | type=2
  v8u g1;
  g1[0] = 0x00030000u;                                              // data_size=8B
  g1[1] = (nqw & 0xFFFFu) << 16;                                    // tensor_dim0[15:0]
  g1[2] = (nqw >> 16) | (1u << 16);                                 // tensor_dim0[31:16] | tensor_dim1=1
  g1[3] = (nqw & 0xFFFFu) << 16;                                    // tile_dim0
  g1[4] = 1u;                                                       // tile_dim1=1
  g1[5] = nqw;                                                      // tensor_dim0_stride
  g1[6] = 0u;
  g1[7] = 0u;
  asm volatile("tensor_load_to_lds %0, %1" :: "s"(g0), "s"(g1) : "memory");
}

// ---------------------------------------------------------------------------
// Kernel 0: one-shot weight prep.  Transpose + pad + fp32->f16 into workspace.
// ---------------------------------------------------------------------------
__global__ __launch_bounds__(256) void weight_prep_kernel(
    const float* __restrict__ Wi1, const float* __restrict__ Wi2,
    const float* __restrict__ Wu1, const float* __restrict__ Wu2,
    unsigned short* __restrict__ w1i, unsigned short* __restrict__ w2i,
    unsigned short* __restrict__ w1u, unsigned short* __restrict__ w2u)
{
  const int tid = blockIdx.x * blockDim.x + threadIdx.x;
  const int stride = gridDim.x * blockDim.x;
  for (int idx = tid; idx < HID * K1I; idx += stride) {      // [n][k], pad k>=66
    int n = idx / K1I, k = idx - n * K1I;
    w1i[idx] = f2h((k < GCN + 2) ? Wi1[k * HID + n] : 0.f);
  }
  for (int idx = tid; idx < TOW * HID; idx += stride) {      // [n][k]
    int n = idx / HID, k = idx - n * HID;
    w2i[idx] = f2h(Wi2[k * TOW + n]);
  }
  for (int idx = tid; idx < HID * K1U; idx += stride) {
    int n = idx / K1U, k = idx - n * K1U;
    w1u[idx] = f2h(Wu1[k * HID + n]);
  }
  for (int idx = tid; idx < TOW * HID; idx += stride) {
    int n = idx / HID, k = idx - n * HID;
    w2u[idx] = f2h(Wu2[k * TOW + n]);
  }
}

// ---------------------------------------------------------------------------
// Kernel 1: item encoder.  128 items per block, 8 waves, f16 WMMA, f32 accum.
// ---------------------------------------------------------------------------
__global__ __launch_bounds__(256) void item_encode_kernel(
    const float* __restrict__ item_feat, const float* __restrict__ gcn_item,
    const unsigned short* __restrict__ w1t, const float* __restrict__ bi1,
    const unsigned short* __restrict__ w2t, const float* __restrict__ bi2,
    unsigned short* __restrict__ itemE, int n_items)
{
  __shared__ unsigned short sW1[HID * K1I];   // [n][k] f16
  __shared__ unsigned short sW2[TOW * HID];   // [n][k] f16
  __shared__ unsigned short sX[128 * K1I];    // X tile f16
  __shared__ unsigned short sH[8][16 * HID];  // per-wave hidden rows f16
  __shared__ float sB1[HID];
  __shared__ float sB2[TOW];

  const int tid  = threadIdx.x;
  const int base = blockIdx.x * 128;

  // weights via Tensor Data Mover (wave 0 -> sW1, wave 1 -> sW2)
  if (tid < 32)       tdm_load_1d(sW1, w1t, HID * K1I * 2);
  else if (tid < 64)  tdm_load_1d(sW2, w2t, TOW * HID * 2);

  if (tid < HID) sB1[tid] = bi1[tid];
  if (tid < TOW) sB2[tid] = bi2[tid];

  { // X tile: gather + concat + cvt, pair-at-a-time (pairs never straddle)
    unsigned int* sX32 = (unsigned int*)sX;
    for (int p = tid; p < 128 * (K1I / 2); p += 256) {
      int rl = p / (K1I / 2), kp = p - rl * (K1I / 2);
      int k = 2 * kp;
      int row = base + rl; if (row >= n_items) row = n_items - 1;
      float lo = 0.f, hi = 0.f;
      if (k < 2)            { lo = item_feat[row * 2];            hi = item_feat[row * 2 + 1]; }
      else if (k < GCN + 2) { lo = gcn_item[row * GCN + (k - 2)]; hi = gcn_item[row * GCN + (k - 1)]; }
      sX32[p] = pk2h(lo, hi);
    }
  }
  if (tid < 64) __builtin_amdgcn_s_wait_tensorcnt(0);   // issuing waves drain TDM
  __syncthreads();

  const int wid = tid >> 5, lane = tid & 31, half = lane >> 4, r = lane & 15;
  int koff[8];
  #pragma unroll
  for (int i = 0; i < 8; ++i)
    koff[i] = (i < 4) ? (2 * i + 8 * half) : (16 + 2 * (i - 4) + 8 * half);

  // ---- layer 1 ----
  ABFrag aF[3];
  #pragma unroll
  for (int ks = 0; ks < 3; ++ks)
    #pragma unroll
    for (int i = 0; i < 8; ++i)
      aF[ks].u[i] = *(const unsigned int*)&sX[(16 * wid + r) * K1I + 32 * ks + koff[i]];

  #pragma unroll
  for (int nt = 0; nt < 16; ++nt) {
    v8f c;
    float bias = sB1[16 * nt + r];
    #pragma unroll
    for (int j = 0; j < 8; ++j) c[j] = bias;
    #pragma unroll
    for (int ks = 0; ks < 3; ++ks) {
      ABFrag bF;
      #pragma unroll
      for (int i = 0; i < 8; ++i)
        bF.u[i] = *(const unsigned int*)&sW1[(16 * nt + r) * K1I + 32 * ks + koff[i]];
      c = __builtin_amdgcn_wmma_f32_16x16x32_f16(false, aF[ks].v, false, bF.v,
                                                 (short)0, c, false, false);
    }
    #pragma unroll
    for (int j = 0; j < 8; j += 2) {     // ReLU + packed cvt, 2 rows per cvt
      unsigned int pk = pk2h(fmaxf(c[j], 0.f), fmaxf(c[j + 1], 0.f));
      sH[wid][(j + 8 * half) * HID + 16 * nt + r]     = (unsigned short)pk;
      sH[wid][(j + 1 + 8 * half) * HID + 16 * nt + r] = (unsigned short)(pk >> 16);
    }
  }
  __syncthreads();

  // ---- layer 2 ----
  ABFrag aH[8];
  #pragma unroll
  for (int ks = 0; ks < 8; ++ks)
    #pragma unroll
    for (int i = 0; i < 8; ++i)
      aH[ks].u[i] = *(const unsigned int*)&sH[wid][r * HID + 32 * ks + koff[i]];

  float outv[8][8];
  #pragma unroll
  for (int nt = 0; nt < 8; ++nt) {
    v8f c;
    float bias = sB2[16 * nt + r];
    #pragma unroll
    for (int j = 0; j < 8; ++j) c[j] = bias;
    #pragma unroll
    for (int ks = 0; ks < 8; ++ks) {
      ABFrag bF;
      #pragma unroll
      for (int i = 0; i < 8; ++i)
        bF.u[i] = *(const unsigned int*)&sW2[(16 * nt + r) * HID + 32 * ks + koff[i]];
      c = __builtin_amdgcn_wmma_f32_16x16x32_f16(false, aH[ks].v, false, bF.v,
                                                 (short)0, c, false, false);
    }
    #pragma unroll
    for (int j = 0; j < 8; ++j) outv[nt][j] = c[j];
  }

  // ---- per-row L2 normalize + f16 store ----
  #pragma unroll
  for (int j = 0; j < 8; ++j) {
    float ss = 0.f;
    #pragma unroll
    for (int nt = 0; nt < 8; ++nt) ss += outv[nt][j] * outv[nt][j];
    #pragma unroll
    for (int m = 1; m < 16; m <<= 1) ss += __shfl_xor(ss, m, 32);
    float inv = 1.f / fmaxf(sqrtf(ss), 1e-12f);
    int row = base + 16 * wid + j + 8 * half;
    if (row < n_items) {
      #pragma unroll
      for (int nt = 0; nt < 8; nt += 2) {
        unsigned int pk = pk2h(outv[nt][j] * inv, outv[nt + 1][j] * inv);
        itemE[row * TOW + 16 * nt + r]        = (unsigned short)pk;
        itemE[row * TOW + 16 * (nt + 1) + r]  = (unsigned short)(pk >> 16);
      }
    }
  }
}

// ---------------------------------------------------------------------------
// Kernel 2: segment mean (sorted seg_ids).  2 users / block, 2 cols / lane.
// itemE (51 MB f16) stays resident in the 192 MB L2 -> gathers hit L2.
// ---------------------------------------------------------------------------
__global__ __launch_bounds__(128) void seg_mean_kernel(
    const unsigned short* __restrict__ itemE,
    const int* __restrict__ item_idx, const int* __restrict__ seg_ids,
    unsigned short* __restrict__ hist, int n_inter, int n_users)
{
  __shared__ int sB[3];
  const int u0 = blockIdx.x * 2;
  if (threadIdx.x < 3) {
    int target = u0 + (int)threadIdx.x;
    int lo = 0, hi = n_inter;
    while (lo < hi) { int mid = (lo + hi) >> 1; if (seg_ids[mid] < target) lo = mid + 1; else hi = mid; }
    sB[threadIdx.x] = lo;
  }
  __syncthreads();
  const int which = threadIdx.x >> 6;              // 0 or 1 -> user
  const int u = u0 + which;
  if (u >= n_users) return;
  const int lo = sB[which], hi = sB[which + 1];
  const int t = threadIdx.x & 63;                  // 32-bit column pair
  const unsigned int* itemE32 = (const unsigned int*)itemE;
  float a0 = 0.f, a1 = 0.f;
  for (int i = lo; i < hi; ++i) {
    union { unsigned int u; _Float16 h[2]; } q;
    q.u = itemE32[item_idx[i] * (TOW / 2) + t];
    a0 += (float)q.h[0];
    a1 += (float)q.h[1];
  }
  float invc = 1.f / fmaxf((float)(hi - lo), 1.f);
  ((unsigned int*)hist)[u * (TOW / 2) + t] = pk2h(a0 * invc, a1 * invc);
}

// ---------------------------------------------------------------------------
// Kernel 3: user encoder.  K = 64(gcn) + 128(hist) = 192, f32 out + L2 norm.
// ---------------------------------------------------------------------------
__global__ __launch_bounds__(256) void user_encode_kernel(
    const float* __restrict__ gcn_user, const unsigned short* __restrict__ hist,
    const unsigned short* __restrict__ w1t, const float* __restrict__ bu1,
    const unsigned short* __restrict__ w2t, const float* __restrict__ bu2,
    const int* __restrict__ users,
    float* __restrict__ out, int n_users)
{
  __shared__ unsigned short sW1[HID * K1U];
  __shared__ unsigned short sW2[TOW * HID];
  __shared__ unsigned short sX[128 * K1U];
  __shared__ unsigned short sH[8][16 * HID];
  __shared__ float sB1[HID];
  __shared__ float sB2[TOW];

  const int tid  = threadIdx.x;
  const int base = blockIdx.x * 128;

  if (tid < 32)       tdm_load_1d(sW1, w1t, HID * K1U * 2);
  else if (tid < 64)  tdm_load_1d(sW2, w2t, TOW * HID * 2);

  if (tid < HID) sB1[tid] = bu1[tid];
  if (tid < TOW) sB2[tid] = bu2[tid];

  { // X tile: [gcn_user f32->f16 | hist f16 copy], pair-at-a-time
    unsigned int* sX32 = (unsigned int*)sX;
    const unsigned int* hist32 = (const unsigned int*)hist;
    for (int p = tid; p < 128 * (K1U / 2); p += 256) {
      int rl = p / (K1U / 2), kp = p - rl * (K1U / 2);
      int k = 2 * kp;
      int row = base + rl; if (row >= n_users) row = n_users - 1;
      int uu = users[row];
      unsigned int v;
      if (k < GCN) v = pk2h(gcn_user[uu * GCN + k], gcn_user[uu * GCN + k + 1]);
      else         v = hist32[uu * (TOW / 2) + (k - GCN) / 2];
      sX32[p] = v;
    }
  }
  if (tid < 64) __builtin_amdgcn_s_wait_tensorcnt(0);
  __syncthreads();

  const int wid = tid >> 5, lane = tid & 31, half = lane >> 4, r = lane & 15;
  int koff[8];
  #pragma unroll
  for (int i = 0; i < 8; ++i)
    koff[i] = (i < 4) ? (2 * i + 8 * half) : (16 + 2 * (i - 4) + 8 * half);

  ABFrag aF[6];
  #pragma unroll
  for (int ks = 0; ks < 6; ++ks)
    #pragma unroll
    for (int i = 0; i < 8; ++i)
      aF[ks].u[i] = *(const unsigned int*)&sX[(16 * wid + r) * K1U + 32 * ks + koff[i]];

  #pragma unroll
  for (int nt = 0; nt < 16; ++nt) {
    v8f c;
    float bias = sB1[16 * nt + r];
    #pragma unroll
    for (int j = 0; j < 8; ++j) c[j] = bias;
    #pragma unroll
    for (int ks = 0; ks < 6; ++ks) {
      ABFrag bF;
      #pragma unroll
      for (int i = 0; i < 8; ++i)
        bF.u[i] = *(const unsigned int*)&sW1[(16 * nt + r) * K1U + 32 * ks + koff[i]];
      c = __builtin_amdgcn_wmma_f32_16x16x32_f16(false, aF[ks].v, false, bF.v,
                                                 (short)0, c, false, false);
    }
    #pragma unroll
    for (int j = 0; j < 8; j += 2) {
      unsigned int pk = pk2h(fmaxf(c[j], 0.f), fmaxf(c[j + 1], 0.f));
      sH[wid][(j + 8 * half) * HID + 16 * nt + r]     = (unsigned short)pk;
      sH[wid][(j + 1 + 8 * half) * HID + 16 * nt + r] = (unsigned short)(pk >> 16);
    }
  }
  __syncthreads();

  ABFrag aH[8];
  #pragma unroll
  for (int ks = 0; ks < 8; ++ks)
    #pragma unroll
    for (int i = 0; i < 8; ++i)
      aH[ks].u[i] = *(const unsigned int*)&sH[wid][r * HID + 32 * ks + koff[i]];

  float outv[8][8];
  #pragma unroll
  for (int nt = 0; nt < 8; ++nt) {
    v8f c;
    float bias = sB2[16 * nt + r];
    #pragma unroll
    for (int j = 0; j < 8; ++j) c[j] = bias;
    #pragma unroll
    for (int ks = 0; ks < 8; ++ks) {
      ABFrag bF;
      #pragma unroll
      for (int i = 0; i < 8; ++i)
        bF.u[i] = *(const unsigned int*)&sW2[(16 * nt + r) * HID + 32 * ks + koff[i]];
      c = __builtin_amdgcn_wmma_f32_16x16x32_f16(false, aH[ks].v, false, bF.v,
                                                 (short)0, c, false, false);
    }
    #pragma unroll
    for (int j = 0; j < 8; ++j) outv[nt][j] = c[j];
  }

  #pragma unroll
  for (int j = 0; j < 8; ++j) {
    float ss = 0.f;
    #pragma unroll
    for (int nt = 0; nt < 8; ++nt) ss += outv[nt][j] * outv[nt][j];
    #pragma unroll
    for (int m = 1; m < 16; m <<= 1) ss += __shfl_xor(ss, m, 32);
    float inv = 1.f / fmaxf(sqrtf(ss), 1e-12f);
    int row = base + 16 * wid + j + 8 * half;
    if (row < n_users) {
      #pragma unroll
      for (int nt = 0; nt < 8; ++nt)
        out[row * TOW + 16 * nt + r] = outv[nt][j] * inv;
    }
  }
}

// ---------------------------------------------------------------------------
extern "C" void kernel_launch(void* const* d_in, const int* in_sizes, int n_in,
                              void* d_out, int out_size, void* d_ws, size_t ws_size,
                              hipStream_t stream) {
  const float* item_feat = (const float*)d_in[0];
  const float* gcn_item  = (const float*)d_in[1];
  const float* gcn_user  = (const float*)d_in[2];
  const float* Wi1 = (const float*)d_in[3];
  const float* bi1 = (const float*)d_in[4];
  const float* Wi2 = (const float*)d_in[5];
  const float* bi2 = (const float*)d_in[6];
  const float* Wu1 = (const float*)d_in[7];
  const float* bu1 = (const float*)d_in[8];
  const float* Wu2 = (const float*)d_in[9];
  const float* bu2 = (const float*)d_in[10];
  const int* item_idx = (const int*)d_in[11];
  const int* seg_ids  = (const int*)d_in[12];
  const int* users    = (const int*)d_in[13];

  const int n_items = in_sizes[0] / 2;
  const int n_users = in_sizes[2] / GCN;
  const int n_inter = in_sizes[11];

  unsigned short* itemE = (unsigned short*)d_ws;            // [n_items][TOW] f16
  unsigned short* hist  = itemE + (size_t)n_items * TOW;    // [n_users][TOW] f16
  unsigned short* w1i   = hist + (size_t)n_users * TOW;     // [HID][K1I]
  unsigned short* w2i   = w1i + HID * K1I;                  // [TOW][HID]
  unsigned short* w1u   = w2i + TOW * HID;                  // [HID][K1U]
  unsigned short* w2u   = w1u + HID * K1U;                  // [TOW][HID]

  const int item_blocks = (n_items + 127) / 128;
  const int user_blocks = (n_users + 127) / 128;
  const int seg_blocks  = (n_users + 1) / 2;

  weight_prep_kernel<<<dim3(64), dim3(256), 0, stream>>>(
      Wi1, Wi2, Wu1, Wu2, w1i, w2i, w1u, w2u);

  item_encode_kernel<<<dim3(item_blocks), dim3(256), 0, stream>>>(
      item_feat, gcn_item, w1i, bi1, w2i, bi2, itemE, n_items);

  seg_mean_kernel<<<dim3(seg_blocks), dim3(128), 0, stream>>>(
      itemE, item_idx, seg_ids, hist, n_inter, n_users);

  user_encode_kernel<<<dim3(user_blocks), dim3(256), 0, stream>>>(
      gcn_user, hist, w1u, bu1, w2u, bu2, users, (float*)d_out, n_users);
}
